// MultiHeadAttention_50019189129223
// MI455X (gfx1250) — compile-verified, matrix operand order checked by reading, not run on hardware
//
#include <hip/hip_runtime.h>

#define Dm   1024
#define Hh   16
#define DH   64
#define Bsz  2
#define Lseq 2048
#define Mtot (Bsz * Lseq)   // 4096

typedef __bf16 bf8   __attribute__((ext_vector_type(8)));
typedef __bf16 bf16v __attribute__((ext_vector_type(16)));
typedef float  f8    __attribute__((ext_vector_type(8)));

__device__ __forceinline__ bf16v mkfrag(const __bf16* p0, const __bf16* p1) {
    bf8 lo = *(const bf8*)p0;
    bf8 hi = *(const bf8*)p1;
    return __builtin_shufflevector(lo, hi, 0,1,2,3,4,5,6,7,8,9,10,11,12,13,14,15);
}

__device__ __forceinline__ f8 wmma_bf16(bf16v a, bf16v b, f8 c) {
    return __builtin_amdgcn_wmma_f32_16x16x32_bf16(false, a, false, b, (short)0, c, false, false);
}

// ---------------- f32 -> bf16 convert ----------------
__global__ void cvt_f32_bf16(const float* __restrict__ s, __bf16* __restrict__ d, int n) {
    int i = blockIdx.x * blockDim.x + threadIdx.x;
    int stride = gridDim.x * blockDim.x;
    for (; i < n; i += stride) d[i] = (__bf16)s[i];
}

// ---------------- GEMM: C[row,col] = sum_k A[row,k]*W[col,k] + bias[col] ----
// mode 0: dst bf16, per-head layout  [B,H,L,DH]
// mode 1: dst bf16, transposed head  [B,H,DH,L]   (V^T for attention B-operand)
// mode 2: dst f32,  flat [Mtot, Dm]  (final output projection)
__global__ __launch_bounds__(256)
void gemm16(const __bf16* __restrict__ A, const __bf16* __restrict__ W,
            const float* __restrict__ bias, void* __restrict__ dst, int mode) {
    const int lane = threadIdx.x & 31;
    const int wave = threadIdx.x >> 5;
    const int r    = lane & 15;
    const int half = lane >> 4;
    const int m0 = blockIdx.x * 128 + wave * 16;
    const int n0 = blockIdx.y * 64;

    f8 c0 = {}, c1 = {}, c2 = {}, c3 = {};
    const __bf16* arow = A + (size_t)(m0 + r) * Dm;
    const __bf16* w0   = W + (size_t)(n0 + r) * Dm;
    const __bf16* w1   = w0 + (size_t)16 * Dm;
    const __bf16* w2   = w0 + (size_t)32 * Dm;
    const __bf16* w3   = w0 + (size_t)48 * Dm;

    for (int kk = 0; kk < Dm; kk += 32) {
        bf16v af = mkfrag(arow + kk + half*8, arow + kk + 16 + half*8);
        bf16v b0 = mkfrag(w0 + kk + half*8, w0 + kk + 16 + half*8);
        bf16v b1 = mkfrag(w1 + kk + half*8, w1 + kk + 16 + half*8);
        bf16v b2 = mkfrag(w2 + kk + half*8, w2 + kk + 16 + half*8);
        bf16v b3 = mkfrag(w3 + kk + half*8, w3 + kk + 16 + half*8);
        c0 = wmma_bf16(af, b0, c0);
        c1 = wmma_bf16(af, b1, c1);
        c2 = wmma_bf16(af, b2, c2);
        c3 = wmma_bf16(af, b3, c3);
    }

    f8 cc[4] = {c0, c1, c2, c3};
    #pragma unroll
    for (int t = 0; t < 4; ++t) {
        int col = n0 + t * 16 + r;
        float bv = bias[col];
        #pragma unroll
        for (int jj = 0; jj < 8; ++jj) {
            int row = m0 + half * 8 + jj;
            float val = cc[t][jj] + bv;
            if (mode == 2) {
                ((float*)dst)[(size_t)row * Dm + col] = val;
            } else {
                int b = row >> 11, l = row & (Lseq - 1);
                int h = col >> 6,  dh = col & (DH - 1);
                if (mode == 0)
                    ((__bf16*)dst)[((size_t)(b * Hh + h) * Lseq + l) * DH + dh] = (__bf16)val;
                else
                    ((__bf16*)dst)[((size_t)(b * Hh + h) * DH + dh) * Lseq + l] = (__bf16)val;
            }
        }
    }
}

// ---------------- Flash attention: per-wave 16 query rows --------------------
__global__ __launch_bounds__(128)
void flash16(const __bf16* __restrict__ qh, const __bf16* __restrict__ kh,
             const __bf16* __restrict__ vT, __bf16* __restrict__ attnout) {
    const int lane = threadIdx.x & 31;
    const int wave = threadIdx.x >> 5;
    const int r    = lane & 15;
    const int half = lane >> 4;
    const int qblk = blockIdx.x & 31;           // L/64 = 32 query blocks
    const int bh   = blockIdx.x >> 5;           // B*H = 32
    const int b    = bh >> 4;
    const int h    = bh & 15;
    const int m0   = qblk * 64 + wave * 16;

    const __bf16* qb = qh + (size_t)bh * Lseq * DH;
    const __bf16* kb = kh + (size_t)bh * Lseq * DH;
    const __bf16* vb = vT + (size_t)bh * DH * Lseq;

    const __bf16* qrow = qb + (size_t)(m0 + r) * DH;
    bf16v aq0 = mkfrag(qrow + half*8,      qrow + 16 + half*8);
    bf16v aq1 = mkfrag(qrow + 32 + half*8, qrow + 48 + half*8);

    f8 o0 = {}, o1 = {}, o2 = {}, o3 = {};
    float mrow[8], lrow[8];
    #pragma unroll
    for (int jj = 0; jj < 8; ++jj) { mrow[jj] = -3.0e38f; lrow[jj] = 0.0f; }

    __shared__ alignas(16) __bf16 ptile[4][16 * 32];
    __bf16* pt = &ptile[wave][0];

    for (int k0 = 0; k0 < Lseq; k0 += 32) {
        const __bf16* kr0 = kb + (size_t)(k0 + r) * DH;
        const __bf16* kr1 = kb + (size_t)(k0 + 16 + r) * DH;
        bf16v bk00 = mkfrag(kr0 + half*8,      kr0 + 16 + half*8);
        bf16v bk01 = mkfrag(kr0 + 32 + half*8, kr0 + 48 + half*8);
        bf16v bk10 = mkfrag(kr1 + half*8,      kr1 + 16 + half*8);
        bf16v bk11 = mkfrag(kr1 + 32 + half*8, kr1 + 48 + half*8);

        f8 s0 = {}; s0 = wmma_bf16(aq0, bk00, s0); s0 = wmma_bf16(aq1, bk01, s0);
        f8 s1 = {}; s1 = wmma_bf16(aq0, bk10, s1); s1 = wmma_bf16(aq1, bk11, s1);

        #pragma unroll
        for (int jj = 0; jj < 8; ++jj) {
            float a  = s0[jj] * 0.125f;           // 1/sqrt(64)
            float bb = s1[jj] * 0.125f;
            float mx = fmaxf(a, bb);
            mx = fmaxf(mx, __shfl_xor(mx, 1));
            mx = fmaxf(mx, __shfl_xor(mx, 2));
            mx = fmaxf(mx, __shfl_xor(mx, 4));
            mx = fmaxf(mx, __shfl_xor(mx, 8));
            float mn = fmaxf(mrow[jj], mx);
            float al = __expf(mrow[jj] - mn);
            mrow[jj] = mn;
            float p0 = __expf(a - mn);
            float p1 = __expf(bb - mn);
            float sum = p0 + p1;
            sum += __shfl_xor(sum, 1);
            sum += __shfl_xor(sum, 2);
            sum += __shfl_xor(sum, 4);
            sum += __shfl_xor(sum, 8);
            lrow[jj] = lrow[jj] * al + sum;
            o0[jj] *= al; o1[jj] *= al; o2[jj] *= al; o3[jj] *= al;
            pt[(half*8 + jj) * 32 + r]      = (__bf16)p0;   // C-layout -> LDS
            pt[(half*8 + jj) * 32 + 16 + r] = (__bf16)p1;
        }

        asm volatile("s_wait_dscnt 0x0" ::: "memory");      // P writes visible

        bf16v pf = mkfrag(pt + r*32 + half*8, pt + r*32 + 16 + half*8); // A-layout
        const __bf16* vr = vb + (size_t)r * Lseq + k0;
        bf16v v0 = mkfrag(vr + half*8,             vr + 16 + half*8);
        bf16v v1 = mkfrag(vr + 16*Lseq + half*8,   vr + 16*Lseq + 16 + half*8);
        bf16v v2 = mkfrag(vr + 32*Lseq + half*8,   vr + 32*Lseq + 16 + half*8);
        bf16v v3 = mkfrag(vr + 48*Lseq + half*8,   vr + 48*Lseq + 16 + half*8);
        o0 = wmma_bf16(pf, v0, o0);
        o1 = wmma_bf16(pf, v1, o1);
        o2 = wmma_bf16(pf, v2, o2);
        o3 = wmma_bf16(pf, v3, o3);
    }

    #pragma unroll
    for (int jj = 0; jj < 8; ++jj) {
        float inv = 1.0f / lrow[jj];
        size_t rowbase = (size_t)(b * Lseq + m0 + half*8 + jj) * Dm + h * DH;
        attnout[rowbase +      r] = (__bf16)(o0[jj] * inv);
        attnout[rowbase + 16 + r] = (__bf16)(o1[jj] * inv);
        attnout[rowbase + 32 + r] = (__bf16)(o2[jj] * inv);
        attnout[rowbase + 48 + r] = (__bf16)(o3[jj] * inv);
    }
}

extern "C" void kernel_launch(void* const* d_in, const int* in_sizes, int n_in,
                              void* d_out, int out_size, void* d_ws, size_t ws_size,
                              hipStream_t stream) {
    const float* Q  = (const float*)d_in[0];
    const float* K  = (const float*)d_in[1];
    const float* V  = (const float*)d_in[2];
    const float* Wq = (const float*)d_in[3];
    const float* bq = (const float*)d_in[4];
    const float* Wk = (const float*)d_in[5];
    const float* bk = (const float*)d_in[6];
    const float* Wv = (const float*)d_in[7];
    const float* bv = (const float*)d_in[8];
    const float* Wo = (const float*)d_in[9];
    const float* bo = (const float*)d_in[10];

    char* ws = (char*)d_ws;
    const size_t MB = 1024 * 1024;
    __bf16* Qbf = (__bf16*)(ws + 0 * MB);
    __bf16* Kbf = (__bf16*)(ws + 8 * MB);
    __bf16* Vbf = (__bf16*)(ws + 16 * MB);
    __bf16* Wqb = (__bf16*)(ws + 24 * MB);
    __bf16* Wkb = (__bf16*)(ws + 26 * MB);
    __bf16* Wvb = (__bf16*)(ws + 28 * MB);
    __bf16* Wob = (__bf16*)(ws + 30 * MB);
    __bf16* qh  = (__bf16*)(ws + 32 * MB);
    __bf16* khd = (__bf16*)(ws + 40 * MB);
    __bf16* vT  = (__bf16*)(ws + 48 * MB);
    __bf16* ao  = (__bf16*)(ws + 56 * MB);

    const int NTOK = Mtot * Dm;   // 4M elements each for Q/K/V
    const int NW   = Dm * Dm;     // 1M per weight matrix

    cvt_f32_bf16<<<2048, 256, 0, stream>>>(Q,  Qbf, NTOK);
    cvt_f32_bf16<<<2048, 256, 0, stream>>>(K,  Kbf, NTOK);
    cvt_f32_bf16<<<2048, 256, 0, stream>>>(V,  Vbf, NTOK);
    cvt_f32_bf16<<<1024, 256, 0, stream>>>(Wq, Wqb, NW);
    cvt_f32_bf16<<<1024, 256, 0, stream>>>(Wk, Wkb, NW);
    cvt_f32_bf16<<<1024, 256, 0, stream>>>(Wv, Wvb, NW);
    cvt_f32_bf16<<<1024, 256, 0, stream>>>(Wo, Wob, NW);

    dim3 g(Mtot / 128, Dm / 64), blk(256);
    gemm16<<<g, blk, 0, stream>>>(Qbf, Wqb, bq, (void*)qh,  0);
    gemm16<<<g, blk, 0, stream>>>(Kbf, Wkb, bk, (void*)khd, 0);
    gemm16<<<g, blk, 0, stream>>>(Vbf, Wvb, bv, (void*)vT,  1);

    flash16<<<Bsz * Hh * (Lseq / 64), 128, 0, stream>>>(qh, khd, vT, ao);

    gemm16<<<g, blk, 0, stream>>>(ao, Wob, bo, d_out, 2);
}
